// TrambaUltra_70231305224197
// MI455X (gfx1250) — matrix operand order repaired and checked by reading, not compiled, not run on hardware
//
#include <hip/hip_runtime.h>
#include <math.h>

// ---------------- model dims ----------------
#define NB    8
#define NPTS  4096
#define NG    512
#define KNN   32
#define ENCF  256
#define BD    384
#define NHEAD 6
#define HD    64
#define LSEQ  512
#define DI    768
#define DS    16
#define DTR   24
#define CKW   4
#define SEG   50

typedef __attribute__((ext_vector_type(16))) _Float16 v16h;
typedef __attribute__((ext_vector_type(8)))  _Float16 v8h;
typedef __attribute__((ext_vector_type(8)))  float    v8f;

// ---------------- math helpers ----------------
__device__ __forceinline__ float sigf(float x){ return 1.f/(1.f+__expf(-x)); }
__device__ __forceinline__ float geluf(float x){
  float x3 = x*x*x;
  return 0.5f*x*(1.f+tanhf(0.7978845608f*(x+0.044715f*x3)));
}
__device__ __forceinline__ float splusf(float x){
  return (x > 20.f) ? x : log1pf(__expf(x));
}

// ---------------- WMMA fragment loaders (LDS f16 -> VGPR) ----------------
// A operand (16x32 f16): lane M = lane%16, kbase = (lane/16)*8, halves at +0..7 and +16..23
__device__ __forceinline__ v16h frag_a(const _Float16* rowp, int lane){
  int kb = (lane >> 4) << 3;
  v8h lo = *(const v8h*)(rowp + kb);
  v8h hi = *(const v8h*)(rowp + 16 + kb);
  return __builtin_shufflevector(lo, hi, 0,1,2,3,4,5,6,7,8,9,10,11,12,13,14,15);
}
// B operand (32x16 f16, stored transposed [N][K] in LDS): lane N = lane%16, kbase = (lane/16)*8, halves at +0..7 and +16..23
__device__ __forceinline__ v16h frag_b(const _Float16* rowp, int lane){
  int kb = (lane >> 4) << 3;
  v8h lo = *(const v8h*)(rowp + kb);
  v8h hi = *(const v8h*)(rowp + 16 + kb);
  return __builtin_shufflevector(lo, hi, 0,1,2,3,4,5,6,7,8,9,10,11,12,13,14,15);
}

#define WMMA_F16(A, B, C) \
  __builtin_amdgcn_wmma_f32_16x16x32_f16(false, (A), false, (B), (short)0, (C), false, false)

// ---------------- generic WMMA GEMM: C = act(A@W + bias) + residual ----------------
// A: MxK (row stride lda, f32), W: KxN row-major f32, C/res: MxN f32
// Block 256 threads = 8 waves -> 128x64 tile; each wave owns a 32x32 patch (2x2 WMMA accums).
// K staged in chunks of 32 via LDS f16; next chunk prefetched into cache.
#define ACT_NONE 0
#define ACT_RELU 1
#define ACT_GELU 2
#define ACT_SIGM 3
#define ACT_SPLS 4

template<int ACT>
__global__ __launch_bounds__(256) void gemm_kernel(
    const float* __restrict__ A, int lda,
    const float* __restrict__ W,
    const float* __restrict__ bias,
    const float* __restrict__ res,
    float* __restrict__ C,
    int M, int N, int K)
{
  __shared__ __align__(16) _Float16 As[128][32];
  __shared__ __align__(16) _Float16 Bs[64][32];
  const int tid  = threadIdx.x;
  const int row0 = blockIdx.y * 128;
  const int col0 = blockIdx.x * 64;
  const int wave = tid >> 5, lane = tid & 31;
  const int wm = wave >> 1, wn = wave & 1;   // 4x2 waves of 32x32 patches
  const int lm = lane & 15;
  v8f a00 = {}, a01 = {}, a10 = {}, a11 = {};

  for (int k0 = 0; k0 < K; k0 += 32) {
    // prefetch next K-chunk into cache (global_prefetch_b8)
    if (k0 + 32 < K) {
      if (tid < 128 && row0 + tid < M)
        __builtin_prefetch(&A[(size_t)(row0 + tid)*lda + k0 + 32], 0, 3);
      int pkk = tid >> 3, pnn = (tid & 7) * 8;
      if (k0 + 32 + pkk < K && col0 + pnn < N)
        __builtin_prefetch(&W[(size_t)(k0 + 32 + pkk)*N + col0 + pnn], 0, 3);
    }
    #pragma unroll
    for (int r = 0; r < 16; ++r) {                    // 128x32 A tile
      int i = tid + 256*r;
      int m = i >> 5, kk = i & 31;
      int gr = row0 + m, gk = k0 + kk;
      float v = (gr < M && gk < K) ? A[(size_t)gr*lda + gk] : 0.f;
      As[m][kk] = (_Float16)v;
    }
    #pragma unroll
    for (int r = 0; r < 8; ++r) {                     // 64x32 W^T tile
      int i = tid + 256*r;
      int kk = i >> 6, n = i & 63;
      int gk = k0 + kk, gc = col0 + n;
      float v = (gk < K && gc < N) ? W[(size_t)gk*N + gc] : 0.f;
      Bs[n][kk] = (_Float16)v;
    }
    __syncthreads();
    v16h fa0 = frag_a(&As[wm*32 + lm][0],      lane);
    v16h fa1 = frag_a(&As[wm*32 + 16 + lm][0], lane);
    v16h fb0 = frag_b(&Bs[wn*32 + lm][0],      lane);
    v16h fb1 = frag_b(&Bs[wn*32 + 16 + lm][0], lane);
    a00 = WMMA_F16(fa0, fb0, a00);
    a01 = WMMA_F16(fa0, fb1, a01);
    a10 = WMMA_F16(fa1, fb0, a10);
    a11 = WMMA_F16(fa1, fb1, a11);
    __syncthreads();
  }

  const int rhalf = (lane >> 4) << 3;
  #pragma unroll
  for (int i = 0; i < 2; ++i) {
    #pragma unroll
    for (int j = 0; j < 2; ++j) {
      const v8f acc = (i==0) ? (j==0 ? a00 : a01) : (j==0 ? a10 : a11);
      const int col = col0 + wn*32 + j*16 + lm;
      if (col < N) {
        float bv = bias ? bias[col] : 0.f;
        #pragma unroll
        for (int v = 0; v < 8; ++v) {
          int row = row0 + wm*32 + i*16 + rhalf + v;
          if (row < M) {
            float val = acc[v] + bv;
            if      (ACT == ACT_RELU) val = fmaxf(val, 0.f);
            else if (ACT == ACT_GELU) val = geluf(val);
            else if (ACT == ACT_SIGM) val = sigf(val);
            else if (ACT == ACT_SPLS) val = splusf(val);
            if (res) val += res[(size_t)row*N + col];
            C[(size_t)row*N + col] = val;
          }
        }
      }
    }
  }
}

// ---------------- flash attention (WMMA): per (b,h,32-row q tile) ----------------
// qkv rows [B*L][1152] = [q(384)|k(384)|v(384)], head slice h*64..h*64+63
__global__ __launch_bounds__(128) void attn_kernel(const float* __restrict__ qkv,
                                                   float* __restrict__ out)
{
  __shared__ __align__(16) _Float16 Qs[32][64];
  __shared__ __align__(16) _Float16 Ks[32][64];
  __shared__ __align__(16) _Float16 Vs[64][32];     // transposed V: Vs[d][j]
  __shared__ float Ssh[32][33];
  __shared__ __align__(16) _Float16 Psh[32][32];
  __shared__ float mrow[32], lrow[32], alphas[32];

  const int tid = threadIdx.x;
  const int wave = tid >> 5, lane = tid & 31, lm = lane & 15;
  const int q0 = blockIdx.x * 32;
  const int h  = blockIdx.y;
  const int b  = blockIdx.z;
  const size_t rowbase = (size_t)b * LSEQ;

  // load Q tile (scaled by 1/sqrt(64))
  for (int i = tid; i < 32*64; i += 128) {
    int r = i >> 6, d = i & 63;
    Qs[r][d] = (_Float16)(0.125f * qkv[(rowbase + q0 + r)*1152 + h*HD + d]);
  }
  if (tid < 32) { mrow[tid] = -1e30f; lrow[tid] = 0.f; }
  __syncthreads();

  v8f o0 = {}, o1 = {};                     // wave covers out cols [wave*16, wave*16+16)
  const int tm = wave >> 1, tn = wave & 1;  // S-tile assignment (2x2 of 16x16)
  const int rhalf = (lane >> 4) << 3;

  for (int kb = 0; kb < LSEQ; kb += 32) {
    for (int i = tid; i < 32*64; i += 128) {
      int r = i >> 6, d = i & 63;
      float kvk = qkv[(rowbase + kb + r)*1152 + 384 + h*HD + d];
      float kvv = qkv[(rowbase + kb + r)*1152 + 768 + h*HD + d];
      Ks[r][d] = (_Float16)kvk;
      Vs[d][r] = (_Float16)kvv;
    }
    __syncthreads();

    // S tile = Q @ K^T  (K-dim = 64 -> two chunks of 32)
    v8f s = {};
    const _Float16* qp = &Qs[tm*16 + lm][0];
    const _Float16* kp = &Ks[tn*16 + lm][0];
    s = WMMA_F16(frag_a(qp, lane),    frag_b(kp, lane),    s);
    s = WMMA_F16(frag_a(qp+32, lane), frag_b(kp+32, lane), s);
    #pragma unroll
    for (int v = 0; v < 8; ++v)
      Ssh[tm*16 + rhalf + v][tn*16 + lm] = s[v];
    __syncthreads();

    // online softmax per row
    if (tid < 32) {
      int r = tid;
      float mb = -1e30f;
      #pragma unroll 4
      for (int j = 0; j < 32; ++j) mb = fmaxf(mb, Ssh[r][j]);
      float mo = mrow[r];
      float mn = fmaxf(mo, mb);
      float al = __expf(mo - mn);
      float sum = 0.f;
      #pragma unroll 4
      for (int j = 0; j < 32; ++j) {
        float p = __expf(Ssh[r][j] - mn);
        Psh[r][j] = (_Float16)p;
        sum += p;
      }
      lrow[r] = lrow[r]*al + sum;
      mrow[r] = mn;
      alphas[r] = al;
    }
    __syncthreads();

    // rescale O, then O += P @ V
    #pragma unroll
    for (int v = 0; v < 8; ++v) { o0[v] *= alphas[rhalf + v]; o1[v] *= alphas[16 + rhalf + v]; }
    v16h bv = frag_b(&Vs[wave*16 + lm][0], lane);
    o0 = WMMA_F16(frag_a(&Psh[lm][0],    lane), bv, o0);
    o1 = WMMA_F16(frag_a(&Psh[16+lm][0], lane), bv, o1);
    __syncthreads();
  }

  const int col = h*HD + wave*16 + lm;
  #pragma unroll
  for (int v = 0; v < 8; ++v) {
    int r0 = rhalf + v, r1 = 16 + rhalf + v;
    out[(rowbase + q0 + r0)*BD + col] = o0[v] / lrow[r0];
    out[(rowbase + q0 + r1)*BD + col] = o1[v] / lrow[r1];
  }
}

// ---------------- KNN: block per (b,g), 256 threads ----------------
__global__ __launch_bounds__(256) void knn_kernel(const float* __restrict__ pts, int* __restrict__ idx)
{
  __shared__ float dist[NPTS];
  __shared__ float rv[256];
  __shared__ int   ri[256];
  const int tid = threadIdx.x;
  const int bg = blockIdx.x;
  const int b = bg / NG, g = bg % NG;
  const float* pb = pts + (size_t)b * NPTS * 3;
  const float cx = pb[(g*8)*3+0], cy = pb[(g*8)*3+1], cz = pb[(g*8)*3+2];
  for (int i = tid; i < NPTS; i += 256) {
    float dx = pb[i*3+0]-cx, dy = pb[i*3+1]-cy, dz = pb[i*3+2]-cz;
    dist[i] = dx*dx + dy*dy + dz*dz;
  }
  __syncthreads();
  for (int k = 0; k < KNN; ++k) {
    float bvv = 1e30f; int bi = 0;
    #pragma unroll 4
    for (int j = 0; j < 16; ++j) {
      int i = tid*16 + j;
      float d = dist[i];
      if (d < bvv || (d == bvv && i < bi)) { bvv = d; bi = i; }
    }
    rv[tid] = bvv; ri[tid] = bi;
    __syncthreads();
    for (int s = 128; s > 0; s >>= 1) {
      if (tid < s) {
        if (rv[tid+s] < rv[tid] || (rv[tid+s] == rv[tid] && ri[tid+s] < ri[tid])) {
          rv[tid] = rv[tid+s]; ri[tid] = ri[tid+s];
        }
      }
      __syncthreads();
    }
    if (tid == 0) { idx[(size_t)bg*KNN + k] = ri[0]; dist[ri[0]] = 1e30f; }
    __syncthreads();
  }
}

// ---------------- group feature: feat = max_k relu(rel @ w1 + b1) ----------------
__global__ __launch_bounds__(256) void group_feat_kernel(const float* __restrict__ pts,
    const int* __restrict__ idx, const float* __restrict__ w1, const float* __restrict__ b1,
    float* __restrict__ feat)
{
  __shared__ float rel[KNN][3];
  const int tid = threadIdx.x;
  const int bg = blockIdx.x;
  const int b = bg / NG, g = bg % NG;
  const float* pb = pts + (size_t)b * NPTS * 3;
  if (tid < KNN) {
    int n = idx[(size_t)bg*KNN + tid];
    rel[tid][0] = pb[n*3+0] - pb[(g*8)*3+0];
    rel[tid][1] = pb[n*3+1] - pb[(g*8)*3+1];
    rel[tid][2] = pb[n*3+2] - pb[(g*8)*3+2];
  }
  __syncthreads();
  const int e = tid;  // 256 channels
  float w0 = w1[0*ENCF + e], w1v = w1[1*ENCF + e], w2v = w1[2*ENCF + e], bb = b1[e];
  float m = -1e30f;
  #pragma unroll 4
  for (int j = 0; j < KNN; ++j) {
    float f = fmaxf(rel[j][0]*w0 + rel[j][1]*w1v + rel[j][2]*w2v + bb, 0.f);
    m = fmaxf(m, f);
  }
  feat[(size_t)bg*ENCF + e] = m;
}

// ---------------- pos embed stage1: gelu(centers @ w1 + b1), 4096x128 ----------------
__global__ __launch_bounds__(128) void pos1_kernel(const float* __restrict__ pts,
    const float* __restrict__ w, const float* __restrict__ bb, float* __restrict__ out)
{
  const int row = blockIdx.x, e = threadIdx.x;
  const int b = row / NG, g = row % NG;
  const float* c = pts + ((size_t)b * NPTS + g*8) * 3;
  float v = c[0]*w[0*128+e] + c[1]*w[1*128+e] + c[2]*w[2*128+e] + bb[e];
  out[(size_t)row*128 + e] = geluf(v);
}

// ---------------- layernorm: wave per row (D=384) ----------------
__global__ __launch_bounds__(256) void ln_kernel(const float* __restrict__ x,
    const float* __restrict__ g, const float* __restrict__ bta, float* __restrict__ y, int rows)
{
  const int wave = threadIdx.x >> 5, lane = threadIdx.x & 31;
  const int row = blockIdx.x*8 + wave;
  if (row >= rows) return;
  const float* xr = x + (size_t)row*BD;
  float v[12], s = 0.f, s2 = 0.f;
  #pragma unroll
  for (int i = 0; i < 12; ++i) { v[i] = xr[lane + 32*i]; s += v[i]; s2 += v[i]*v[i]; }
  #pragma unroll
  for (int off = 16; off >= 1; off >>= 1) { s += __shfl_xor(s, off); s2 += __shfl_xor(s2, off); }
  float mean = s * (1.f/BD);
  float var  = s2 * (1.f/BD) - mean*mean;
  float inv  = rsqrtf(var + 1e-5f);
  float* yr = y + (size_t)row*BD;
  #pragma unroll
  for (int i = 0; i < 12; ++i) {
    int c = lane + 32*i;
    yr[c] = (v[i]-mean)*inv*g[c] + bta[c];
  }
}

// ---------------- score = s1 @ w2 + b2 (N=1), wave per row ----------------
__global__ __launch_bounds__(256) void score_kernel(const float* __restrict__ s1,
    const float* __restrict__ w2, const float* __restrict__ b2, float* __restrict__ sc, int rows)
{
  const int wave = threadIdx.x >> 5, lane = threadIdx.x & 31;
  const int row = blockIdx.x*8 + wave;
  if (row >= rows) return;
  float s = 0.f;
  #pragma unroll
  for (int i = 0; i < 12; ++i) { int c = lane + 32*i; s += s1[(size_t)row*BD + c]*w2[c]; }
  #pragma unroll
  for (int off = 16; off >= 1; off >>= 1) s += __shfl_xor(s, off);
  if (lane == 0) sc[row] = s + b2[0];
}

// ---------------- bitonic argsort (desc + asc, stable via index tiebreak) ----------------
__global__ __launch_bounds__(256) void sort_kernel(const float* __restrict__ score,
    int* __restrict__ idxd, int* __restrict__ idxa)
{
  __shared__ float ks[LSEQ];
  __shared__ int   vs[LSEQ];
  const int b = blockIdx.x, tid = threadIdx.x;
  for (int pass = 0; pass < 2; ++pass) {
    for (int i = tid; i < LSEQ; i += 256) { ks[i] = score[b*LSEQ + i]; vs[i] = i; }
    __syncthreads();
    for (int k = 2; k <= LSEQ; k <<= 1) {
      for (int j = k >> 1; j > 0; j >>= 1) {
        for (int t = tid; t < LSEQ; t += 256) {
          int ixj = t ^ j;
          if (ixj > t) {
            float a = ks[t], c = ks[ixj]; int ia = vs[t], ic = vs[ixj];
            bool tLess = pass ? ((a < c) || (a == c && ia < ic))     // ascending
                              : ((a > c) || (a == c && ia < ic));    // descending
            bool up = ((t & k) == 0);
            bool swap = up ? !tLess : tLess;
            if (swap) { ks[t] = c; ks[ixj] = a; vs[t] = ic; vs[ixj] = ia; }
          }
        }
        __syncthreads();
      }
    }
    int* dst = pass ? idxa : idxd;
    for (int i = tid; i < LSEQ; i += 256) dst[b*LSEQ + i] = vs[i];
    __syncthreads();
  }
}

// ---------------- gather stacked: rows 0..4095 = xd, 4096..8191 = xa ----------------
__global__ __launch_bounds__(256) void gather_kernel(const float* __restrict__ x,
    const int* __restrict__ idxd, const int* __restrict__ idxa, float* __restrict__ xs)
{
  size_t gid = (size_t)blockIdx.x*256 + threadIdx.x;
  if (gid >= (size_t)2*NB*LSEQ*BD) return;
  int row = (int)(gid / BD), c = (int)(gid % BD);
  int s = row / LSEQ, l = row % LSEQ;
  int b = s & 7;
  int src = (s < NB) ? idxd[b*LSEQ + l] : idxa[b*LSEQ + l];
  xs[gid] = x[((size_t)(b*LSEQ + src))*BD + c];
}

// ---------------- depthwise causal conv (CK=4) + silu ----------------
__global__ __launch_bounds__(768) void dwconv_kernel(const float* __restrict__ xz,
    const float* __restrict__ w, const float* __restrict__ b, float* __restrict__ out)
{
  const int row = blockIdx.x, c = threadIdx.x;
  const int s = row >> 9, t = row & 511;
  float acc = b[c];
  #pragma unroll
  for (int k = 0; k < CKW; ++k) {
    int ti = t + k - (CKW-1);
    if (ti >= 0) acc += xz[((size_t)(s*LSEQ + ti))*(2*DI) + c] * w[c*CKW + k];
  }
  acc = acc * sigf(acc);
  out[(size_t)row*DI + c] = acc;
}

// ---------------- selective scan: block per sequence, thread per channel ----------------
__global__ __launch_bounds__(768) void scan_kernel(const float* __restrict__ u,
    const float* __restrict__ dt, const float* __restrict__ xdbl,
    const float* __restrict__ A_log, const float* __restrict__ Dp, float* __restrict__ y)
{
  __shared__ float BC[2*DS];
  const int s = blockIdx.x, d = threadIdx.x;
  float Arow[DS], h[DS];
  #pragma unroll
  for (int j = 0; j < DS; ++j) { Arow[j] = -__expf(A_log[d*DS + j]); h[j] = 0.f; }
  const float Dpd = Dp[d];
  for (int t = 0; t < LSEQ; ++t) {
    size_t row = (size_t)s*LSEQ + t;
    if (d < 2*DS) BC[d] = xdbl[row*56 + DTR + d];
    __syncthreads();
    float dtv = dt[row*DI + d];
    float uv  = u[row*DI + d];
    float acc = 0.f;
    #pragma unroll
    for (int j = 0; j < DS; ++j) {
      h[j] = h[j]*__expf(dtv*Arow[j]) + dtv*BC[j]*uv;
      acc += h[j]*BC[DS + j];
    }
    y[row*DI + d] = acc + uv*Dpd;
    __syncthreads();
  }
}

// ---------------- y *= silu(z)  (z = xz cols 768..1535) ----------------
__global__ __launch_bounds__(256) void ymul_kernel(float* __restrict__ y, const float* __restrict__ xz)
{
  size_t gid = (size_t)blockIdx.x*256 + threadIdx.x;
  if (gid >= (size_t)2*NB*LSEQ*DI) return;
  size_t row = gid / DI, c = gid % DI;
  float z = xz[row*(2*DI) + DI + c];
  y[gid] *= z * sigf(z);
}

// ---------------- concat [xd|xa] -> 4096x768 ----------------
__global__ __launch_bounds__(256) void concat_kernel(const float* __restrict__ xs, float* __restrict__ out)
{
  size_t gid = (size_t)blockIdx.x*256 + threadIdx.x;
  if (gid >= (size_t)NB*LSEQ*2*BD) return;
  size_t row = gid / (2*BD); int c = (int)(gid % (2*BD));
  float v = (c < BD) ? xs[row*BD + c] : xs[(row + (size_t)NB*LSEQ)*BD + (c - BD)];
  out[gid] = v;
}

// ---------------- x = g*xd + (1-g)*xa ----------------
__global__ __launch_bounds__(256) void fusegate_kernel(const float* __restrict__ g,
    const float* __restrict__ xs, float* __restrict__ x)
{
  size_t gid = (size_t)blockIdx.x*256 + threadIdx.x;
  if (gid >= (size_t)NB*LSEQ*BD) return;
  float gg = g[gid];
  float xd = xs[gid];
  float xa = xs[gid + (size_t)NB*LSEQ*BD];
  x[gid] = gg*xd + (1.f - gg)*xa;
}

// ============================ host launch ============================
static inline dim3 gemm_grid(int M, int N) { return dim3((N + 63)/64, (M + 127)/128); }

extern "C" void kernel_launch(void* const* d_in, const int* in_sizes, int n_in,
                              void* d_out, int out_size, void* d_ws, size_t ws_size,
                              hipStream_t stream) {
  (void)in_sizes; (void)n_in; (void)out_size; (void)ws_size;
  auto F = [&](int i) { return (const float*)d_in[i]; };

  // leaf order: pts, grouping/pos params, trf[6]x12, imp x4, mamba[6]x13, fuse/norm/head
  const float* pts = F(0);
  const int IN_GRP_W1=1, IN_GRP_B1=2, IN_GRP_W2=3, IN_GRP_B2=4;
  const int IN_POS_W1=5, IN_POS_B1=6, IN_POS_W2=7, IN_POS_B2=8;
  const int IN_TRF0=9;     // +0 ln1_g +1 ln1_b +2 Wqkv +3 bqkv +4 Wo +5 bo +6 ln2_g +7 ln2_b +8 Wm1 +9 bm1 +10 Wm2 +11 bm2
  const int IN_IMP_W1=81, IN_IMP_B1=82, IN_IMP_W2=83, IN_IMP_B2=84;
  const int IN_MAMBA0=85;  // +0 ln_g +1 ln_b +2 W_in +3 b_in +4 conv_w +5 conv_b +6 W_x +7 W_dt +8 b_dt +9 A_log +10 Dp +11 W_out +12 b_out
  const int IN_FUSE_W=163, IN_FUSE_B=164, IN_NORM_G=165, IN_NORM_B=166;
  const int IN_HEAD_W1=167, IN_HEAD_B1=168, IN_HEAD_W2=169, IN_HEAD_B2=170;

  const int Mx  = NB*LSEQ;      // 4096 rows (single stream)
  const int Mx2 = 2*NB*LSEQ;    // 8192 rows (stacked xd|xa)

  // workspace layout (floats)
  float* w  = (float*)d_ws;
  int*   iw = (int*)d_ws;
  size_t o = 131072;            // ints region first: knn idx (4096*32), later idxd/idxa
  float* X    = w + o;  o += (size_t)Mx  * BD;       // 4096x384
  float* XN   = w + o;  o += (size_t)Mx2 * BD;       // 8192x384 (LN out)
  float* XS   = w + o;  o += (size_t)Mx2 * BD;       // 8192x384 (stacked stream)
  float* T0   = w + o;  o += (size_t)Mx2 * 1536;     // qkv / mlp-hid / xz
  float* T1   = w + o;  o += (size_t)Mx2 * DI;       // attn-out / xsconv / concat
  float* T2   = w + o;  o += (size_t)Mx2 * DI;       // dt / gate
  float* T3   = w + o;  o += (size_t)Mx2 * DI;       // y (scan out)
  float* XDBL = w + o;  o += (size_t)Mx2 * 56;       // xs @ W_x
  float* SC   = w + o;  o += (size_t)Mx;             // scores
  float* HH   = w + o;  o += (size_t)Mx  * 512;      // head hidden
  int* knn_idx = iw;
  int* idxd = iw;                // reuse after grouping
  int* idxa = iw + Mx;

  // ---- grouping ----
  knn_kernel<<<NB*NG, 256, 0, stream>>>(pts, knn_idx);
  group_feat_kernel<<<NB*NG, 256, 0, stream>>>(pts, knn_idx, F(IN_GRP_W1), F(IN_GRP_B1), T0);
  gemm_kernel<ACT_NONE><<<gemm_grid(Mx, BD), 256, 0, stream>>>(T0, ENCF, F(IN_GRP_W2), F(IN_GRP_B2), nullptr, XN, Mx, BD, ENCF);   // tokens
  pos1_kernel<<<Mx, 128, 0, stream>>>(pts, F(IN_POS_W1), F(IN_POS_B1), T1);
  gemm_kernel<ACT_NONE><<<gemm_grid(Mx, BD), 256, 0, stream>>>(T1, 128, F(IN_POS_W2), F(IN_POS_B2), XN, X, Mx, BD, 128);           // x = tokens + pos

  const int lnBlocks  = (Mx  + 7)/8;
  const int lnBlocks2 = (Mx2 + 7)/8;

  // ---- transformer x6 ----
  for (int l = 0; l < 6; ++l) {
    const int p = IN_TRF0 + 12*l;
    ln_kernel<<<lnBlocks, 256, 0, stream>>>(X, F(p+0), F(p+1), XN, Mx);
    gemm_kernel<ACT_NONE><<<gemm_grid(Mx, 3*BD), 256, 0, stream>>>(XN, BD, F(p+2), F(p+3), nullptr, T0, Mx, 3*BD, BD);  // qkv
    attn_kernel<<<dim3(LSEQ/32, NHEAD, NB), 128, 0, stream>>>(T0, T1);
    gemm_kernel<ACT_NONE><<<gemm_grid(Mx, BD), 256, 0, stream>>>(T1, BD, F(p+4), F(p+5), X, X, Mx, BD, BD);             // proj + residual
    ln_kernel<<<lnBlocks, 256, 0, stream>>>(X, F(p+6), F(p+7), XN, Mx);
    gemm_kernel<ACT_GELU><<<gemm_grid(Mx, 4*BD), 256, 0, stream>>>(XN, BD, F(p+8), F(p+9), nullptr, T0, Mx, 4*BD, BD);  // mlp up
    gemm_kernel<ACT_NONE><<<gemm_grid(Mx, BD), 256, 0, stream>>>(T0, 4*BD, F(p+10), F(p+11), X, X, Mx, BD, 4*BD);       // mlp down + residual
  }

  // ---- importance + sort + gather ----
  gemm_kernel<ACT_RELU><<<gemm_grid(Mx, BD), 256, 0, stream>>>(X, BD, F(IN_IMP_W1), F(IN_IMP_B1), nullptr, XN, Mx, BD, BD);
  score_kernel<<<lnBlocks, 256, 0, stream>>>(XN, F(IN_IMP_W2), F(IN_IMP_B2), SC, Mx);
  sort_kernel<<<NB, 256, 0, stream>>>(SC, idxd, idxa);
  gather_kernel<<<(int)(((size_t)Mx2*BD + 255)/256), 256, 0, stream>>>(X, idxd, idxa, XS);

  // ---- mamba x6 on stacked (xd|xa) ----
  for (int l = 0; l < 6; ++l) {
    const int p = IN_MAMBA0 + 13*l;
    ln_kernel<<<lnBlocks2, 256, 0, stream>>>(XS, F(p+0), F(p+1), XN, Mx2);
    gemm_kernel<ACT_NONE><<<gemm_grid(Mx2, 2*DI), 256, 0, stream>>>(XN, BD, F(p+2), F(p+3), nullptr, T0, Mx2, 2*DI, BD);   // xz
    dwconv_kernel<<<Mx2, DI, 0, stream>>>(T0, F(p+4), F(p+5), T1);                                                          // xs conv+silu
    gemm_kernel<ACT_NONE><<<gemm_grid(Mx2, 56), 256, 0, stream>>>(T1, DI, F(p+6), nullptr, nullptr, XDBL, Mx2, 56, DI);    // xdbl
    gemm_kernel<ACT_SPLS><<<gemm_grid(Mx2, DI), 256, 0, stream>>>(XDBL, 56, F(p+7), F(p+8), nullptr, T2, Mx2, DI, DTR);    // dt
    scan_kernel<<<2*NB, DI, 0, stream>>>(T1, T2, XDBL, F(p+9), F(p+10), T3);                                                // selective scan
    ymul_kernel<<<(int)(((size_t)Mx2*DI + 255)/256), 256, 0, stream>>>(T3, T0);                                             // y *= silu(z)
    gemm_kernel<ACT_NONE><<<gemm_grid(Mx2, BD), 256, 0, stream>>>(T3, DI, F(p+11), F(p+12), XS, XS, Mx2, BD, DI);          // out + residual
  }

  // ---- fuse + head ----
  concat_kernel<<<(int)(((size_t)Mx*2*BD + 255)/256), 256, 0, stream>>>(XS, T1);
  gemm_kernel<ACT_SIGM><<<gemm_grid(Mx, BD), 256, 0, stream>>>(T1, 2*BD, F(IN_FUSE_W), F(IN_FUSE_B), nullptr, T2, Mx, BD, 2*BD);
  fusegate_kernel<<<(int)(((size_t)Mx*BD + 255)/256), 256, 0, stream>>>(T2, XS, X);
  ln_kernel<<<lnBlocks, 256, 0, stream>>>(X, F(IN_NORM_G), F(IN_NORM_B), XN, Mx);
  gemm_kernel<ACT_RELU><<<gemm_grid(Mx, 512), 256, 0, stream>>>(XN, BD, F(IN_HEAD_W1), F(IN_HEAD_B1), nullptr, HH, Mx, 512, BD);
  gemm_kernel<ACT_NONE><<<gemm_grid(Mx, SEG), 256, 0, stream>>>(HH, 512, F(IN_HEAD_W2), F(IN_HEAD_B2), nullptr, (float*)d_out, Mx, SEG, 512);
}
